// KTPBlock_60215441489987
// MI455X (gfx1250) — compile-verified, matrix operand order checked by reading, not run on hardware
//
#include <hip/hip_runtime.h>
#include <hip/hip_fp16.h>
#include <cstdint>

typedef __attribute__((ext_vector_type(16))) _Float16 v16h;
typedef __attribute__((ext_vector_type(8)))  _Float16 v8h;
typedef __attribute__((ext_vector_type(8)))  float    v8f;
typedef __attribute__((ext_vector_type(4)))  unsigned int u32x4;
typedef __attribute__((ext_vector_type(8)))  int      i32x8;
typedef __attribute__((ext_vector_type(4)))  int      i32x4;

// Problem constants (from reference setup_inputs)
#define BB   4
#define NT   1568
#define CC   768
#define HH   3072
#define NH   12
#define SS   196
#define NS   1372     /* NT - SS */
#define NKEEP 902     /* ceil(0.7*1568) - 196 */
#define NCAT 1098     /* SS + NKEEP */
#define M1   (BB*NT)    /* 6272 */
#define M2   (BB*NCAT)  /* 4392 */
#define QTILES (NT/16)  /* 98 */
#define XOUT ((size_t)BB*NCAT*CC) /* 3373056 floats of x in d_out */

// ---------------- WMMA helpers (CDNA5 16x16x32 f16 -> f32) ----------------

__device__ __forceinline__ v8f wmma_f16(v16h a, v16h b, v8f c) {
  return __builtin_amdgcn_wmma_f32_16x16x32_f16(false, a, false, b, (short)0, c,
                                                false, false);
}

// ds_swizzle xor-shuffle: offset encoding (group-of-32) = and 0x1F | or 0 | xor MASK
template <int MASK>
__device__ __forceinline__ float swz_xor(float x) {
  int r = __builtin_amdgcn_ds_swizzle(__float_as_int(x), 0x7C00 | MASK);
  return __int_as_float(r);
}

// A-fragment (also used for column-major B): 16-bit A 16x32 layout:
// lane L holds row (L&15); halves 0-7 -> K = k0 + (L>=16?8:0) + i
//                halves 8-15 -> K = k0 + 16 + (L>=16?8:0) + i
__device__ __forceinline__ v16h load_frag_rm(const _Float16* __restrict__ base,
                                             int row0, int ld, int k0, int maxRow) {
  int lane = threadIdx.x & 31;
  int r = row0 + (lane & 15);
  if (r > maxRow) r = maxRow;
  int kb = k0 + ((lane >> 4) << 3);
  const _Float16* p = base + (size_t)r * ld + kb;
  v8h lo = *(const v8h*)p;
  v8h hi = *(const v8h*)(p + 16);
  v16h out;
#pragma unroll
  for (int i = 0; i < 8; ++i) { out[i] = lo[i]; out[i + 8] = hi[i]; }
  return out;
}

__device__ __forceinline__ v16h load_frag_lds(const _Float16* base, int row0,
                                              int ld, int k0) {
  int lane = threadIdx.x & 31;
  int r = row0 + (lane & 15);
  int kb = k0 + ((lane >> 4) << 3);
  const _Float16* p = base + r * ld + kb;
  v8h lo = *(const v8h*)p;
  v8h hi = *(const v8h*)(p + 16);
  v16h out;
#pragma unroll
  for (int i = 0; i < 8; ++i) { out[i] = lo[i]; out[i + 8] = hi[i]; }
  return out;
}

// -------- Tensor Data Mover: async-load a 64x32-half tile (rows of a
// row-major [N x K] f16 matrix) into LDS offset 0, padded to 80B row stride.
// D# layout per CDNA5 ISA. --------

__device__ __forceinline__ void tdm_load_tile_b(const _Float16* gsrc, int K,
                                                unsigned lds_off) {
  unsigned long long ga = (unsigned long long)(uintptr_t)gsrc;
  u32x4 g0;
  g0[0] = 1u;                                   // count=1, user descriptor
  g0[1] = lds_off;                              // LDS byte address
  g0[2] = (unsigned)(ga & 0xFFFFFFFFu);         // global_addr[31:0]
  g0[3] = (unsigned)((ga >> 32) & 0x01FFFFFFu)  // global_addr[56:32]
          | 0x80000000u;                        // type=2 ("image")
  i32x8 g1;
  // wg_mask=0 | data_size=1 (2B) | pad_enable | pad_interval=3 (16 DWORDs)
  // | pad_amount=3 (4 DWORDs) -> LDS row stride 64+16 = 80 bytes
  g1[0] = (1 << 16) | (1 << 20) | (3 << 22) | (3 << 25);
  unsigned td0 = (unsigned)K;                   // tensor_dim0 (elements)
  unsigned td1 = 1u << 20;                      // tensor_dim1 (large)
  g1[1] = (int)((td0 & 0xFFFFu) << 16);         // [63:48] = dim0 lo16
  g1[2] = (int)((td0 >> 16) | ((td1 & 0xFFFFu) << 16)); // dim0 hi | dim1 lo
  g1[3] = (int)((td1 >> 16) | (32u << 16));     // dim1 hi | tile_dim0 = 32
  g1[4] = 64;                                   // tile_dim1 = 64, tile_dim2 = 0
  unsigned long long stride = (unsigned long long)(unsigned)K;  // dim0 stride
  g1[5] = (int)(stride & 0xFFFFFFFFu);
  g1[6] = (int)((stride >> 32) & 0xFFFFu);      // dim0 stride hi | dim1 stride lo=0
  g1[7] = 0;
  i32x4 z = {0, 0, 0, 0};
#if __clang_major__ >= 23
  i32x8 z8 = {0, 0, 0, 0, 0, 0, 0, 0};
  __builtin_amdgcn_tensor_load_to_lds(g0, g1, z, z, z8, 0);
#else
  __builtin_amdgcn_tensor_load_to_lds(g0, g1, z, z, 0);
#endif
}

// ---------------- Weight f32 -> f16 conversion ----------------

__global__ __launch_bounds__(256) void cvt_weights_k(
    const float* __restrict__ w0, const float* __restrict__ w1,
    const float* __restrict__ w2, const float* __restrict__ w3,
    _Float16* __restrict__ o0, _Float16* __restrict__ o1,
    _Float16* __restrict__ o2, _Float16* __restrict__ o3) {
  size_t i = (size_t)blockIdx.x * 256 + threadIdx.x;
  const size_t n0 = (size_t)3 * CC * CC, n1 = (size_t)CC * CC,
               n2 = (size_t)HH * CC, n3 = (size_t)CC * HH;
  if (i < n0) o0[i] = (_Float16)w0[i];
  else if (i < n0 + n1) o1[i - n0] = (_Float16)w1[i - n0];
  else if (i < n0 + n1 + n2) o2[i - n0 - n1] = (_Float16)w2[i - n0 - n1];
  else if (i < n0 + n1 + n2 + n3) o3[i - n0 - n1 - n2] = (_Float16)w3[i - n0 - n1 - n2];
}

// ---------------- LayerNorm (f32 in -> f16 out) ----------------

__global__ __launch_bounds__(256) void ln_f16_kernel(
    const float* __restrict__ x, const float* __restrict__ s,
    const float* __restrict__ bb, _Float16* __restrict__ out, int C) {
  int row = blockIdx.x;
  const float* xr = x + (size_t)row * C;
  float sum = 0.f, sq = 0.f;
  for (int i = threadIdx.x; i < C; i += 256) { float v = xr[i]; sum += v; sq += v * v; }
#pragma unroll
  for (int off = 16; off >= 1; off >>= 1) {
    sum += __shfl_xor(sum, off, 32);
    sq  += __shfl_xor(sq, off, 32);
  }
  __shared__ float rs[8], rq[8];
  int w = threadIdx.x >> 5;
  if ((threadIdx.x & 31) == 0) { rs[w] = sum; rq[w] = sq; }
  __syncthreads();
  if (threadIdx.x == 0) {
    float ts = 0.f, tq = 0.f;
    for (int i = 0; i < 8; ++i) { ts += rs[i]; tq += rq[i]; }
    float mu = ts / C;
    float var = tq / C - mu * mu;
    rs[0] = mu;
    rq[0] = rsqrtf(var + 1e-5f);
  }
  __syncthreads();
  float mu = rs[0], rstd = rq[0];
  for (int i = threadIdx.x; i < C; i += 256)
    out[(size_t)row * C + i] = (_Float16)((xr[i] - mu) * rstd * s[i] + bb[i]);
}

// ---------------- Generic WMMA GEMM: out = A[MxK] * W[NxK]^T ----------------
// Block: 128 threads = 4 waves; wave computes 16x64; block tile 64x64.
// B tile (64 cols x 32 k) DMA'd into LDS by the Tensor Data Mover (wave 0),
// published with s_wait_tensorcnt + workgroup barrier, shared by all 4 waves.
// MODE 0: QKV epilogue (+q_bias/v_bias, scale q by D^-0.5, route to q/k/vT f16)
// MODE 1: proj  epilogue (+bias +residual(x), f32 out)
// MODE 2: fc1   epilogue (+bias, exact-erf GELU, f16 out)
// MODE 3: fc2   epilogue (+bias +residual(xcat), f32 out -> d_out)

template <int MODE>
__global__ __launch_bounds__(128) void gemm_wmma_k(
    const _Float16* __restrict__ A, const _Float16* __restrict__ W,
    int M, int N, int K,
    const float* __restrict__ bias, const float* __restrict__ bias2,
    float* __restrict__ outF, _Float16* __restrict__ outH,
    _Float16* __restrict__ qh, _Float16* __restrict__ kh,
    _Float16* __restrict__ vT) {
  __shared__ __align__(16) _Float16 Bs[64 * 40];
  const int lane = threadIdx.x & 31;
  const int wave = threadIdx.x >> 5;
  const int n0 = blockIdx.x * 64;
  const int m0 = blockIdx.y * 64 + wave * 16;
  const int nl = lane & 15;
  const int hiq = lane >> 4;
  v8f acc[4] = {};
  for (int k0 = 0; k0 < K; k0 += 32) {
    __syncthreads();
    if (wave == 0) {
      tdm_load_tile_b(W + (size_t)n0 * K + k0, K, 0u);
      __builtin_amdgcn_s_wait_tensorcnt(0);
    }
    __syncthreads();
    if (k0 + 32 < K)
      __builtin_prefetch(A + (size_t)(m0 + nl) * K + k0 + 32, 0, 3);
    v16h a = load_frag_rm(A, m0, K, k0, M - 1);
#pragma unroll
    for (int t = 0; t < 4; ++t) {
      v16h b = load_frag_lds(Bs, t * 16, 40, 0);
      acc[t] = wmma_f16(a, b, acc[t]);
    }
  }
#pragma unroll
  for (int t = 0; t < 4; ++t) {
    int n = n0 + t * 16 + nl;
#pragma unroll
    for (int r = 0; r < 8; ++r) {
      int m = m0 + r + hiq * 8;
      if (m >= M) continue;
      float val = acc[t][r];
      if (MODE == 0) {
        int b = m / NT, tok = m - b * NT;
        if (n < CC) {                     // q (+bias, *scale)
          val += bias[n];
          int h = n >> 6, d = n & 63;
          qh[(((size_t)b * NH + h) * NT + tok) * 64 + d] = (_Float16)(val * 0.125f);
        } else if (n < 2 * CC) {          // k (no bias)
          int nn = n - CC;
          int h = nn >> 6, d = nn & 63;
          kh[(((size_t)b * NH + h) * NT + tok) * 64 + d] = (_Float16)val;
        } else {                          // v (+bias) stored transposed [d][tok]
          int nn = n - 2 * CC;
          val += bias2[nn];
          int h = nn >> 6, d = nn & 63;
          vT[(((size_t)b * NH + h) * 64 + d) * NT + tok] = (_Float16)val;
        }
      } else if (MODE == 2) {
        val += bias[n];
        float g = 0.5f * val * (1.0f + erff(val * 0.70710678118f));
        outH[(size_t)m * N + n] = (_Float16)g;
      } else {  // MODE 1 and 3: +bias +residual, f32 out
        val += bias[n] + bias2[(size_t)m * N + n];
        outF[(size_t)m * N + n] = val;
      }
    }
  }
}

// ---------------- Flash attention (1 wave per 16-query tile) ----------------

__global__ __launch_bounds__(32) void attn_flash_k(
    const _Float16* __restrict__ qh, const _Float16* __restrict__ kh,
    const _Float16* __restrict__ vT, _Float16* __restrict__ attn_h,
    float* __restrict__ stats) {
  int bh = blockIdx.x / QTILES;
  int qt = blockIdx.x % QTILES;
  int b = bh / NH, h = bh - b * NH;
  const int lane = threadIdx.x & 31;
  const int nl = lane & 15, hiq = lane >> 4;
  const _Float16* qbase = qh + (((size_t)bh * NT) + qt * 16) * 64;
  const _Float16* kbase = kh + ((size_t)bh * NT) * 64;
  const _Float16* vbase = vT + ((size_t)bh * 64) * NT;
  __shared__ __align__(16) _Float16 Ps[16 * 32];
  v16h a0 = load_frag_rm(qbase, 0, 64, 0, 15);
  v16h a1 = load_frag_rm(qbase, 0, 64, 32, 15);
  float rowm[8], rowl[8];
#pragma unroll
  for (int r = 0; r < 8; ++r) { rowm[r] = -3.0e38f; rowl[r] = 0.f; }
  v8f o[4] = {};
  for (int kt = 0; kt < NT; kt += 32) {
    v16h b00 = load_frag_rm(kbase, kt, 64, 0, NT - 1);
    v16h b01 = load_frag_rm(kbase, kt, 64, 32, NT - 1);
    v16h b10 = load_frag_rm(kbase, kt + 16, 64, 0, NT - 1);
    v16h b11 = load_frag_rm(kbase, kt + 16, 64, 32, NT - 1);
    v8f c0 = {}; c0 = wmma_f16(a0, b00, c0); c0 = wmma_f16(a1, b01, c0);
    v8f c1 = {}; c1 = wmma_f16(a0, b10, c1); c1 = wmma_f16(a1, b11, c1);
    __syncthreads();
#pragma unroll
    for (int r = 0; r < 8; ++r) {
      float tm = fmaxf(c0[r], c1[r]);
      tm = fmaxf(tm, swz_xor<8>(tm));
      tm = fmaxf(tm, swz_xor<4>(tm));
      tm = fmaxf(tm, swz_xor<2>(tm));
      tm = fmaxf(tm, swz_xor<1>(tm));
      float nm = fmaxf(rowm[r], tm);
      float al = __expf(rowm[r] - nm);
      float p0 = __expf(c0[r] - nm);
      float p1 = __expf(c1[r] - nm);
      float rssum = p0 + p1;
      rssum += swz_xor<8>(rssum);
      rssum += swz_xor<4>(rssum);
      rssum += swz_xor<2>(rssum);
      rssum += swz_xor<1>(rssum);
      rowl[r] = rowl[r] * al + rssum;
      rowm[r] = nm;
      int rowi = r + hiq * 8;
      Ps[rowi * 32 + nl] = (_Float16)p0;
      Ps[rowi * 32 + 16 + nl] = (_Float16)p1;
#pragma unroll
      for (int t = 0; t < 4; ++t) o[t][r] *= al;
    }
    __syncthreads();
    v16h pf = load_frag_lds(Ps, 0, 32, 0);
#pragma unroll
    for (int t = 0; t < 4; ++t) {
      v16h vf = load_frag_rm(vbase, t * 16, NT, kt, 63);
      o[t] = wmma_f16(pf, vf, o[t]);
    }
  }
  float inv[8];
#pragma unroll
  for (int r = 0; r < 8; ++r) inv[r] = 1.0f / rowl[r];
#pragma unroll
  for (int t = 0; t < 4; ++t) {
#pragma unroll
    for (int r = 0; r < 8; ++r) {
      int tok = qt * 16 + r + hiq * 8;
      attn_h[(((size_t)b * NT) + tok) * CC + h * 64 + t * 16 + nl] =
          (_Float16)(o[t][r] * inv[r]);
    }
  }
  if (nl == 0) {
#pragma unroll
    for (int r = 0; r < 8; ++r) {
      size_t si = ((size_t)bh * NT + qt * 16 + r + hiq * 8) * 2;
      stats[si] = rowm[r];
      stats[si + 1] = rowl[r];
    }
  }
}

// ------------- Column-score pass: recompute P with saved (m,l) -------------

__global__ __launch_bounds__(32) void attn_score_k(
    const _Float16* __restrict__ qh, const _Float16* __restrict__ kh,
    const float* __restrict__ stats, float* __restrict__ score) {
  int bh = blockIdx.x / QTILES, qt = blockIdx.x % QTILES;
  int b = bh / NH;
  const int lane = threadIdx.x & 31;
  const int nl = lane & 15, hiq = lane >> 4;
  const _Float16* qbase = qh + (((size_t)bh * NT) + qt * 16) * 64;
  const _Float16* kbase = kh + ((size_t)bh * NT) * 64;
  v16h a0 = load_frag_rm(qbase, 0, 64, 0, 15);
  v16h a1 = load_frag_rm(qbase, 0, 64, 32, 15);
  float coef[8], mr[8];
#pragma unroll
  for (int r = 0; r < 8; ++r) {
    int q = qt * 16 + r + hiq * 8;
    size_t si = ((size_t)bh * NT + q) * 2;
    mr[r] = stats[si];
    float l = stats[si + 1];
    float w = (q < SS) ? 2.0f : 1.0f;
    coef[r] = w / (l * ((float)NH * (float)NT));
  }
  for (int kt = 192; kt < NT; kt += 16) {  // first col >= S=196 lives in tile 192
    v16h b0 = load_frag_rm(kbase, kt, 64, 0, NT - 1);
    v16h b1 = load_frag_rm(kbase, kt, 64, 32, NT - 1);
    v8f c = {};
    c = wmma_f16(a0, b0, c);
    c = wmma_f16(a1, b1, c);
    float cs = 0.f;
#pragma unroll
    for (int r = 0; r < 8; ++r) cs += coef[r] * __expf(c[r] - mr[r]);
    cs += swz_xor<16>(cs);
    int col = kt + nl;
    if (hiq == 0 && col >= SS)
      atomicAdd(&score[(size_t)b * NS + (col - SS)], cs);
  }
}

// ---------------- Top-k via in-LDS bitonic sort (desc, tie: low idx) --------

__global__ __launch_bounds__(1024) void topk_kernel(
    const float* __restrict__ score, const long long* __restrict__ last_idx,
    int* __restrict__ kidx, float* __restrict__ idx_out_dst) {
  __shared__ float ssv[2048];
  __shared__ int siv[2048];
  int b = blockIdx.x;
  for (int i = threadIdx.x; i < 2048; i += 1024) {
    ssv[i] = (i < NS) ? score[(size_t)b * NS + i] : -3.0e38f;
    siv[i] = i;
  }
  __syncthreads();
  for (int k = 2; k <= 2048; k <<= 1) {
    for (int j = k >> 1; j > 0; j >>= 1) {
      for (int i = threadIdx.x; i < 2048; i += 1024) {
        int ixj = i ^ j;
        if (ixj > i) {
          float a = ssv[i], c = ssv[ixj];
          int ia = siv[i], ic = siv[ixj];
          bool aAfter = (a < c) || (a == c && ia > ic);  // a ranks after c (desc)
          bool dirDesc = ((i & k) == 0);
          if (dirDesc ? aAfter : !aAfter) {
            ssv[i] = c; ssv[ixj] = a;
            siv[i] = ic; siv[ixj] = ia;
          }
        }
      }
      __syncthreads();
    }
  }
  if (threadIdx.x < NKEEP) {
    int id = siv[threadIdx.x];
    kidx[b * NKEEP + threadIdx.x] = id;
    idx_out_dst[(size_t)b * NKEEP + threadIdx.x] =
        (float)last_idx[(size_t)b * NS + id];
  }
}

// ---------------- Gather kept tokens ----------------

__global__ __launch_bounds__(256) void gather_rows(
    const float* __restrict__ xa, const int* __restrict__ kidx,
    float* __restrict__ xcat) {
  int row = blockIdx.x;  // 0..M2-1
  int b = row / NCAT, i = row - b * NCAT;
  int src = (i < SS) ? i : (SS + kidx[b * NKEEP + (i - SS)]);
  const float* sp = xa + ((size_t)b * NT + src) * CC;
  float* dp = xcat + (size_t)row * CC;
  for (int j = threadIdx.x; j < CC; j += 256) dp[j] = sp[j];
}

// ---------------- Launch ----------------

extern "C" void kernel_launch(void* const* d_in, const int* in_sizes, int n_in,
                              void* d_out, int out_size, void* d_ws, size_t ws_size,
                              hipStream_t stream) {
  const float* x       = (const float*)d_in[0];
  const long long* last_idx = (const long long*)d_in[1];
  const float* qkv_w   = (const float*)d_in[2];
  const float* q_bias  = (const float*)d_in[3];
  const float* v_bias  = (const float*)d_in[4];
  const float* proj_w  = (const float*)d_in[5];
  const float* proj_b  = (const float*)d_in[6];
  const float* norm1_s = (const float*)d_in[7];
  const float* norm1_b = (const float*)d_in[8];
  const float* norm2_s = (const float*)d_in[9];
  const float* norm2_b = (const float*)d_in[10];
  const float* fc1_w   = (const float*)d_in[11];
  const float* fc1_b   = (const float*)d_in[12];
  const float* fc2_w   = (const float*)d_in[13];
  const float* fc2_b   = (const float*)d_in[14];
  (void)in_sizes; (void)n_in; (void)out_size; (void)ws_size;

  char* ws = (char*)d_ws;
  size_t off = 0;
  auto alloc = [&](size_t bytes) {
    char* p = ws + off;
    off += (bytes + 255) & ~(size_t)255;
    return p;
  };
  _Float16* wh_qkv = (_Float16*)alloc((size_t)3 * CC * CC * 2);
  _Float16* wh_proj = (_Float16*)alloc((size_t)CC * CC * 2);
  _Float16* wh_fc1 = (_Float16*)alloc((size_t)HH * CC * 2);
  _Float16* wh_fc2 = (_Float16*)alloc((size_t)CC * HH * 2);
  _Float16* xn_h   = (_Float16*)alloc((size_t)M1 * CC * 2);
  _Float16* qhp    = (_Float16*)alloc((size_t)BB * NH * NT * 64 * 2);
  _Float16* khp    = (_Float16*)alloc((size_t)BB * NH * NT * 64 * 2);
  _Float16* vTp    = (_Float16*)alloc((size_t)BB * NH * NT * 64 * 2);
  _Float16* attn_h = (_Float16*)alloc((size_t)M1 * CC * 2);
  float*    stats  = (float*)alloc((size_t)BB * NH * NT * 2 * 4);
  float*    score  = (float*)alloc((size_t)BB * NS * 4);
  float*    xa     = (float*)alloc((size_t)M1 * CC * 4);
  int*      kidx   = (int*)alloc((size_t)BB * NKEEP * 4);
  float*    xcat   = (float*)alloc((size_t)M2 * CC * 4);
  _Float16* xc_h   = (_Float16*)alloc((size_t)M2 * CC * 2);
  _Float16* h_h    = (_Float16*)alloc((size_t)M2 * HH * 2);

  hipMemsetAsync(score, 0, (size_t)BB * NS * sizeof(float), stream);

  // weights -> f16 (7,077,888 elems / 256 = 27648 blocks exactly)
  cvt_weights_k<<<27648, 256, 0, stream>>>(qkv_w, proj_w, fc1_w, fc2_w,
                                           wh_qkv, wh_proj, wh_fc1, wh_fc2);
  // LN1
  ln_f16_kernel<<<M1, 256, 0, stream>>>(x, norm1_s, norm1_b, xn_h, CC);
  // QKV GEMM: [6272,768] x [2304,768]^T
  gemm_wmma_k<0><<<dim3(3 * CC / 64, M1 / 64), 128, 0, stream>>>(
      xn_h, wh_qkv, M1, 3 * CC, CC, q_bias, v_bias, nullptr, nullptr,
      qhp, khp, vTp);
  // Flash attention
  attn_flash_k<<<BB * NH * QTILES, 32, 0, stream>>>(qhp, khp, vTp, attn_h, stats);
  // Column scores (recompute with saved stats)
  attn_score_k<<<BB * NH * QTILES, 32, 0, stream>>>(qhp, khp, stats, score);
  // proj GEMM + residual: xa = x + attn @ proj_w^T + proj_b
  gemm_wmma_k<1><<<dim3(CC / 64, M1 / 64), 128, 0, stream>>>(
      attn_h, wh_proj, M1, CC, CC, proj_b, x, xa, nullptr,
      nullptr, nullptr, nullptr);
  // top-k + idx_out (as floats after x in d_out)
  topk_kernel<<<BB, 1024, 0, stream>>>(score, last_idx, kidx,
                                       (float*)d_out + XOUT);
  // gather kept tokens
  gather_rows<<<M2, 256, 0, stream>>>(xa, kidx, xcat);
  // LN2
  ln_f16_kernel<<<M2, 256, 0, stream>>>(xcat, norm2_s, norm2_b, xc_h, CC);
  // FC1 GEMM + exact GELU
  gemm_wmma_k<2><<<dim3(HH / 64, (M2 + 63) / 64), 128, 0, stream>>>(
      xc_h, wh_fc1, M2, HH, CC, fc1_b, nullptr, nullptr, h_h,
      nullptr, nullptr, nullptr);
  // FC2 GEMM + residual -> d_out
  gemm_wmma_k<3><<<dim3(CC / 64, (M2 + 63) / 64), 128, 0, stream>>>(
      h_h, wh_fc2, M2, CC, HH, fc2_b, xcat, (float*)d_out, nullptr,
      nullptr, nullptr, nullptr);
}
